// DynamicEdgeConvNet_44822278701163
// MI455X (gfx1250) — matrix operand order repaired and checked by reading, not compiled, not run on hardware
//
#include <hip/hip_runtime.h>
#include <hip/hip_bf16.h>

// ---------------------------------------------------------------------------
// DynamicEdgeConvNet forward for MI455X (gfx1250, wave32, WMMA bf16).
// - All GEMMs: v_wmma_f32_16x16x32_bf16, weights pre-converted to padded
//   transposed bf16 and streamed to LDS with double-buffered async copies
//   (GLOBAL_LOAD_ASYNC_TO_LDS_B128 when available, guarded by __has_builtin).
// - kNN rewiring: WMMA gram tiles over a pre-converted bf16 node matrix.
// ---------------------------------------------------------------------------

#define NN      16384
#define KNB     20
#define EE      (NN * KNB)      // 327680 edges
#define LAT     128
#define GG      4

typedef __attribute__((ext_vector_type(8)))  float   v8f;
typedef __attribute__((ext_vector_type(16))) __bf16  v16bf;
typedef __attribute__((ext_vector_type(4)))  int     v4i;

#if defined(__has_builtin)
#if __has_builtin(__builtin_amdgcn_global_load_async_to_lds_b128) && \
    __has_builtin(__builtin_amdgcn_s_wait_asynccnt)
#define USE_ASYNC_LDS 1
#endif
#endif
#ifndef USE_ASYNC_LDS
#define USE_ASYNC_LDS 0
#endif

__device__ __forceinline__ void copy16_lds(void* dst, const void* src) {
#if USE_ASYNC_LDS
  __builtin_amdgcn_global_load_async_to_lds_b128(
      (__attribute__((address_space(1))) v4i*)(void*)src,
      (__attribute__((address_space(3))) v4i*)dst, 0, 0);
#else
  *(float4*)dst = *(const float4*)src;
#endif
}

__device__ __forceinline__ void wait_async_lds() {
#if USE_ASYNC_LDS
  __builtin_amdgcn_s_wait_asynccnt(0);
#endif
}

__device__ __forceinline__ float gelu_f(float x) {
  float x3 = x * x * x;
  return 0.5f * x * (1.0f + tanhf(0.7978845608028654f * (x + 0.044715f * x3)));
}

// ---------------------------------------------------------------------------
// Generic GEMM: Y[M x 128] = act(Asrc[M x K] @ W[K x 128] + bias)
//   Wt: pre-converted bf16, transposed+padded: Wt[n * Kpad + k] = W[k][n]
// MODE 0: A = Aplain (row-major, 128 cols)
// MODE 1: A = concat(nodes[snd]-nodes[rcv] (128), g (4))         K=132
// MODE 2: A = concat(prev_edges (128), diff (128), g (4))        K=260
// MODE 3: A = concat(nodes (128), recv (128), g (4))             K=260
// Block = 256 threads = 8 waves; wave w -> rows [blk*128+w*16, +16), all 128
// cols as 8 WMMA accumulators. Weight chunks (32 K) double-buffered in LDS
// via async copies.
// ---------------------------------------------------------------------------
template <int MODE, bool GELU>
__global__ __launch_bounds__(256) void k_gemm(
    const float* __restrict__ Aplain,
    const float* __restrict__ nodesP, const float* __restrict__ recvP,
    const float* __restrict__ gP,
    const int* __restrict__ snd, const int* __restrict__ rcv,
    const __bf16* __restrict__ Wt, const float* __restrict__ bias,
    float* __restrict__ Y, int M, int Kchunks) {
  __shared__ __bf16 Wl[2][128][32];  // Wl[b][n][k] = Wt[n*Kpad + kc*32 + k]

  const int tid  = threadIdx.x;
  const int wave = tid >> 5;
  const int lane = tid & 31;
  const int rowBase = blockIdx.x * 128 + wave * 16;
  const int mrow = rowBase + (lane & 15);   // A-matrix row owned by this lane
  const int Kpad = Kchunks * 32;

  int si = 0, ri = 0;
  if constexpr (MODE == 1 || MODE == 2) { si = snd[mrow]; ri = rcv[mrow]; }
  float g0 = 0.f, g1 = 0.f, g2 = 0.f, g3 = 0.f;
  if constexpr (MODE != 0) { g0 = gP[0]; g1 = gP[1]; g2 = gP[2]; g3 = gP[3]; }

  auto ldA = [&](int k) -> float {
    if constexpr (MODE == 0) {
      return Aplain[(size_t)mrow * 128 + k];
    } else if constexpr (MODE == 1) {
      if (k < 128) return nodesP[(size_t)si * 128 + k] - nodesP[(size_t)ri * 128 + k];
      if (k < 132) return (k == 128) ? g0 : (k == 129) ? g1 : (k == 130) ? g2 : g3;
      return 0.f;
    } else if constexpr (MODE == 2) {
      if (k < 128) return Aplain[(size_t)mrow * 128 + k];
      if (k < 256) { int kk = k - 128;
        return nodesP[(size_t)si * 128 + kk] - nodesP[(size_t)ri * 128 + kk]; }
      if (k < 260) return (k == 256) ? g0 : (k == 257) ? g1 : (k == 258) ? g2 : g3;
      return 0.f;
    } else {  // MODE 3
      if (k < 128) return nodesP[(size_t)mrow * 128 + k];
      if (k < 256) return recvP[(size_t)mrow * 128 + (k - 128)];
      if (k < 260) return (k == 256) ? g0 : (k == 257) ? g1 : (k == 258) ? g2 : g3;
      return 0.f;
    }
  };

  // stage chunk kc into buffer b: 8KB = 512 x 16B, 2 per thread
  auto stage = [&](int kc, int b) {
#pragma unroll
    for (int t = tid; t < 512; t += 256) {
      int n = t >> 2;
      int q = t & 3;
      copy16_lds(&Wl[b][n][q * 8], Wt + (size_t)n * Kpad + kc * 32 + q * 8);
    }
  };

  const v8f zero8 = {0.f, 0.f, 0.f, 0.f, 0.f, 0.f, 0.f, 0.f};
  v8f acc[8];
#pragma unroll
  for (int nt = 0; nt < 8; ++nt) acc[nt] = zero8;

  const int ncol = lane & 15;
  const int kbB  = (lane < 16) ? 0 : 8;

  stage(0, 0);
  for (int kc = 0; kc < Kchunks; ++kc) {
    const int cur = kc & 1;
    wait_async_lds();       // this wave's outstanding chunk copy done
    __syncthreads();        // all waves' copies visible; prev buffer free
    if (kc + 1 < Kchunks) stage(kc + 1, 1 - cur);

    // ---- A fragment (16x32 bf16, documented CDNA5 layout) ----
    const int kb = kc * 32 + ((lane < 16) ? 0 : 8);
    v16bf a;
#pragma unroll
    for (int i = 0; i < 4; ++i) {
      a[2 * i]         = (__bf16)ldA(kb + 2 * i);
      a[2 * i + 1]     = (__bf16)ldA(kb + 2 * i + 1);
      a[8 + 2 * i]     = (__bf16)ldA(kb + 16 + 2 * i);
      a[8 + 2 * i + 1] = (__bf16)ldA(kb + 16 + 2 * i + 1);
    }

    // ---- 8 column tiles, each one WMMA ----
#pragma unroll
    for (int nt = 0; nt < 8; ++nt) {
      v16bf b;
#pragma unroll
      for (int i = 0; i < 4; ++i) {
        b[2 * i]         = Wl[cur][nt * 16 + ncol][kbB + 2 * i];
        b[2 * i + 1]     = Wl[cur][nt * 16 + ncol][kbB + 2 * i + 1];
        b[8 + 2 * i]     = Wl[cur][nt * 16 + ncol][kbB + 16 + 2 * i];
        b[8 + 2 * i + 1] = Wl[cur][nt * 16 + ncol][kbB + 16 + 2 * i + 1];
      }
      acc[nt] = __builtin_amdgcn_wmma_f32_16x16x32_bf16(
          false, a, false, b, (short)0, acc[nt], false, false);
    }
  }

  // ---- epilogue: bias (+gelu), documented C layout ----
  const int moff = (lane < 16) ? 0 : 8;
#pragma unroll
  for (int nt = 0; nt < 8; ++nt) {
    int col = nt * 16 + ncol;
    float bb = bias[col];
#pragma unroll
    for (int r = 0; r < 8; ++r) {
      int m = rowBase + moff + r;
      float y = acc[nt][r] + bb;
      if (GELU) y = gelu_f(y);
      if (m < M) Y[(size_t)m * 128 + col] = y;
    }
  }
}

// ---------------------------------------------------------------------------
// kNN: one wave per 16-row block, WMMA gram tiles vs all 1024 column tiles
// over the pre-converted bf16 node matrix (L2-resident, 4MB), 16 lanes keep
// per-row ascending top-20 lists (matches top_k(-d) stable order).
// ---------------------------------------------------------------------------
__global__ __launch_bounds__(32) void k_knn(
    const __bf16* __restrict__ nb, const float* __restrict__ sqn,
    int* __restrict__ sndO, int* __restrict__ rcvO) {
  __shared__ float tile[16][16];
  __shared__ float bd[16][KNB];
  __shared__ int   bi[16][KNB];
  const int lane = threadIdx.x;
  const int base = blockIdx.x * 16;

  if (lane < 16) {
#pragma unroll
    for (int t = 0; t < KNB; ++t) { bd[lane][t] = 3.4e38f; bi[lane][t] = 0; }
  }

  // A fragments for this row block (fixed for whole scan): 4 chunks of K=32
  const int m   = base + (lane & 15);
  const int kb8 = (lane < 16) ? 0 : 8;
  v16bf a[4];
#pragma unroll
  for (int kc = 0; kc < 4; ++kc) {
#pragma unroll
    for (int i = 0; i < 4; ++i) {
      int k0 = kc * 32 + kb8 + 2 * i;
      a[kc][2 * i]     = nb[(size_t)m * 128 + k0];
      a[kc][2 * i + 1] = nb[(size_t)m * 128 + k0 + 1];
      int k1 = kc * 32 + kb8 + 16 + 2 * i;
      a[kc][8 + 2 * i]     = nb[(size_t)m * 128 + k1];
      a[kc][8 + 2 * i + 1] = nb[(size_t)m * 128 + k1 + 1];
    }
  }

  const int ncol = lane & 15;
  const int kb16 = (lane < 16) ? 0 : 8;
  const int moff = (lane < 16) ? 0 : 8;
  const v8f zero8 = {0.f, 0.f, 0.f, 0.f, 0.f, 0.f, 0.f, 0.f};
  __syncthreads();

  for (int ct = 0; ct < NN / 16; ++ct) {
    const int j = ct * 16 + ncol;     // B column = node index
    v8f accv = zero8;
#pragma unroll
    for (int kc = 0; kc < 4; ++kc) {
      v16bf b;  // B[k][n] = nodes[j][k]  (gram: B = nodes^T)
#pragma unroll
      for (int i = 0; i < 4; ++i) {
        int k0 = kc * 32 + kb16 + 2 * i;
        b[2 * i]     = nb[(size_t)j * 128 + k0];
        b[2 * i + 1] = nb[(size_t)j * 128 + k0 + 1];
        b[8 + 2 * i]     = nb[(size_t)j * 128 + k0 + 16];
        b[8 + 2 * i + 1] = nb[(size_t)j * 128 + k0 + 17];
      }
      accv = __builtin_amdgcn_wmma_f32_16x16x32_bf16(
          false, a[kc], false, b, (short)0, accv, false, false);
    }
    float sj = sqn[j];
#pragma unroll
    for (int r = 0; r < 8; ++r) tile[moff + r][ncol] = sj - 2.0f * accv[r];
    __syncthreads();

    if (lane < 16) {
      for (int n = 0; n < 16; ++n) {
        float s = tile[lane][n];
        if (s < bd[lane][KNB - 1]) {
          int p = KNB - 1;
          while (p > 0 && bd[lane][p - 1] > s) {
            bd[lane][p] = bd[lane][p - 1];
            bi[lane][p] = bi[lane][p - 1];
            --p;
          }
          bd[lane][p] = s;
          bi[lane][p] = ct * 16 + n;
        }
      }
    }
    __syncthreads();
  }

  if (lane < 16) {
    for (int t = 0; t < KNB; ++t) {
      sndO[(size_t)(base + lane) * KNB + t] = base + lane;
      rcvO[(size_t)(base + lane) * KNB + t] = bi[lane][t];
    }
  }
}

// ---------------------------------------------------------------------------
// Small kernels
// ---------------------------------------------------------------------------

// Pre-convert weight matrix to transposed padded bf16: Wt[n*Kpad+k] = W[k][n]
__global__ void k_prepw(const float* __restrict__ W, __bf16* __restrict__ Wt,
                        int Kreal, int Kpad) {
  int gid = blockIdx.x * blockDim.x + threadIdx.x;
  if (gid >= 128 * Kpad) return;
  int n = gid / Kpad, k = gid - n * Kpad;
  Wt[gid] = (k < Kreal) ? (__bf16)W[(size_t)k * 128 + n] : (__bf16)0.f;
}

__global__ void k_tobf16(const float* __restrict__ X, __bf16* __restrict__ Y, int nelem) {
  int gid = blockIdx.x * blockDim.x + threadIdx.x;
  if (gid < nelem) Y[gid] = (__bf16)X[gid];
}

__global__ void k_embed_l1(const float* __restrict__ x, const float* __restrict__ W,
                           const float* __restrict__ b, float* __restrict__ out) {
  int gid = blockIdx.x * blockDim.x + threadIdx.x;
  int row = gid >> 7, c = gid & 127;
  float acc = b[c];
#pragma unroll
  for (int j = 0; j < 3; ++j) acc += x[row * 3 + j] * W[j * 128 + c];
  out[gid] = gelu_f(acc);
}

__global__ void k_dec_final(const float* __restrict__ h, const float* __restrict__ W,
                            const float* __restrict__ b, float* __restrict__ out) {
  int gid = blockIdx.x * blockDim.x + threadIdx.x;
  if (gid >= NN * 3) return;
  int row = gid / 3, c = gid - row * 3;
  float acc = b[c];
  for (int j = 0; j < 128; ++j) acc += h[(size_t)row * 128 + j] * W[j * 3 + c];
  out[gid] = acc;
}

__global__ void k_sqnorm(const float* __restrict__ nodes, float* __restrict__ sqn) {
  int i = blockIdx.x * blockDim.x + threadIdx.x;
  if (i >= NN) return;
  const float* r = nodes + (size_t)i * 128;
  float s = 0.f;
  for (int k = 0; k < 128; ++k) s += r[k] * r[k];
  sqn[i] = s;
}

__global__ void k_scatter_add(float* __restrict__ recvOut, const float* __restrict__ edges,
                              const int* __restrict__ rcv, int E) {
  int gid = blockIdx.x * blockDim.x + threadIdx.x;
  if (gid >= E * 128) return;
  int e = gid >> 7, c = gid & 127;
  atomicAdd(recvOut + (size_t)rcv[e] * 128 + c, edges[gid]);
}

__global__ __launch_bounds__(256) void k_layernorm(
    float* __restrict__ Y, const float* __restrict__ X, const float* __restrict__ res,
    const float* __restrict__ scale, const float* __restrict__ bias, int M) {
  int row  = blockIdx.x * 8 + (threadIdx.x >> 5);
  int lane = threadIdx.x & 31;
  if (row >= M) return;
  float v[4];
#pragma unroll
  for (int i = 0; i < 4; ++i) {
    int c = i * 32 + lane;
    v[i] = X[(size_t)row * 128 + c];
    if (res) v[i] += res[(size_t)row * 128 + c];
  }
  float s = v[0] + v[1] + v[2] + v[3];
#pragma unroll
  for (int off = 16; off > 0; off >>= 1) s += __shfl_xor(s, off, 32);
  float mu = s * (1.0f / 128.0f);
  float q = 0.f;
#pragma unroll
  for (int i = 0; i < 4; ++i) { v[i] -= mu; q += v[i] * v[i]; }
#pragma unroll
  for (int off = 16; off > 0; off >>= 1) q += __shfl_xor(q, off, 32);
  float inv = rsqrtf(q * (1.0f / 128.0f) + 1e-6f);
#pragma unroll
  for (int i = 0; i < 4; ++i) {
    int c = i * 32 + lane;
    Y[(size_t)row * 128 + c] = v[i] * inv * scale[c] + bias[c];
  }
}

// ---------------------------------------------------------------------------
// Host orchestration
// ---------------------------------------------------------------------------
extern "C" void kernel_launch(void* const* d_in, const int* in_sizes, int n_in,
                              void* d_out, int out_size, void* d_ws, size_t ws_size,
                              hipStream_t stream) {
  (void)in_sizes; (void)n_in; (void)out_size; (void)ws_size;

  // ---- inputs (jax pytree flatten: top dict insertion order, params sorted) ----
  const float* x    = (const float*)d_in[0];
  const float* g    = (const float*)d_in[1];
  const int*   snd0 = (const int*)d_in[2];
  const int*   rcv0 = (const int*)d_in[3];
  // params: decoder(4..9), edge0(10..15), edge1(16..21), edge2(22..27),
  //         embed(28..33), ln_bias(34), ln_scale(35), node0..2(36..53)
  const float* decW[3] = {(const float*)d_in[4], (const float*)d_in[6], (const float*)d_in[8]};
  const float* decB[3] = {(const float*)d_in[5], (const float*)d_in[7], (const float*)d_in[9]};
  const float* embW[3] = {(const float*)d_in[28], (const float*)d_in[30], (const float*)d_in[32]};
  const float* embB[3] = {(const float*)d_in[29], (const float*)d_in[31], (const float*)d_in[33]};
  const float* ln_bias  = (const float*)d_in[34];
  const float* ln_scale = (const float*)d_in[35];

  // ---- workspace carve-up ----
  char* p = (char*)d_ws;
  auto carve = [&](size_t bytes) -> void* {
    void* r = (void*)p;
    p += (bytes + 255) & ~(size_t)255;
    return r;
  };
  float*  nodes   = (float*)carve((size_t)NN * 128 * 4);
  float*  recvB   = (float*)carve((size_t)NN * 128 * 4);
  float*  sqn     = (float*)carve((size_t)NN * 4);
  __bf16* nodesBF = (__bf16*)carve((size_t)NN * 128 * 2);
  int*    sndW    = (int*)carve((size_t)EE * 4);
  int*    rcvW    = (int*)carve((size_t)EE * 4);
  float*  t0      = (float*)carve((size_t)EE * 128 * 4);
  float*  t1      = (float*)carve((size_t)EE * 128 * 4);
  float*  edges   = (float*)carve((size_t)EE * 128 * 4);

  // ---- pre-convert all weights: bf16, transposed, K padded to 32 ----
  auto prep = [&](const float* W, int Kreal, int Kpad) -> __bf16* {
    __bf16* wt = (__bf16*)carve((size_t)128 * Kpad * 2);
    int total = 128 * Kpad;
    k_prepw<<<(total + 255) / 256, 256, 0, stream>>>(W, wt, Kreal, Kpad);
    return wt;
  };
  __bf16* wtEmb1 = prep(embW[1], 128, 128);
  __bf16* wtEmb2 = prep(embW[2], 128, 128);
  __bf16* wtDec0 = prep(decW[0], 128, 128);
  __bf16* wtDec1 = prep(decW[1], 128, 128);
  __bf16 *wtE[3][3], *wtN[3][3];
  for (int s = 0; s < 3; ++s) {
    wtE[s][0] = prep((const float*)d_in[10 + 6 * s], s == 0 ? 132 : 260, s == 0 ? 160 : 288);
    wtE[s][1] = prep((const float*)d_in[12 + 6 * s], 128, 128);
    wtE[s][2] = prep((const float*)d_in[14 + 6 * s], 128, 128);
    wtN[s][0] = prep((const float*)d_in[36 + 6 * s], 260, 288);
    wtN[s][1] = prep((const float*)d_in[38 + 6 * s], 128, 128);
    wtN[s][2] = prep((const float*)d_in[40 + 6 * s], 128, 128);
  }

  // ---- encoder ----
  k_embed_l1<<<NN * 128 / 256, 256, 0, stream>>>(x, embW[0], embB[0], t0);
  k_gemm<0, true ><<<NN / 128, 256, 0, stream>>>(t0, nullptr, nullptr, nullptr, nullptr, nullptr,
                                                 wtEmb1, embB[1], t1, NN, 4);
  k_gemm<0, false><<<NN / 128, 256, 0, stream>>>(t1, nullptr, nullptr, nullptr, nullptr, nullptr,
                                                 wtEmb2, embB[2], nodes, NN, 4);

  const int* cs = snd0;
  const int* cr = rcv0;

  for (int step = 0; step < 3; ++step) {
    const float* eB[3] = {(const float*)d_in[11 + 6 * step], (const float*)d_in[13 + 6 * step],
                          (const float*)d_in[15 + 6 * step]};
    const float* nB[3] = {(const float*)d_in[37 + 6 * step], (const float*)d_in[39 + 6 * step],
                          (const float*)d_in[41 + 6 * step]};

    // edge MLP
    if (step == 0)
      k_gemm<1, true><<<EE / 128, 256, 0, stream>>>(nullptr, nodes, nullptr, g, cs, cr,
                                                    wtE[0][0], eB[0], t0, EE, 5);
    else
      k_gemm<2, true><<<EE / 128, 256, 0, stream>>>(edges, nodes, nullptr, g, cs, cr,
                                                    wtE[step][0], eB[0], t0, EE, 9);
    k_gemm<0, true ><<<EE / 128, 256, 0, stream>>>(t0, nullptr, nullptr, nullptr, nullptr, nullptr,
                                                   wtE[step][1], eB[1], t1, EE, 4);
    k_gemm<0, false><<<EE / 128, 256, 0, stream>>>(t1, nullptr, nullptr, nullptr, nullptr, nullptr,
                                                   wtE[step][2], eB[2], edges, EE, 4);

    // aggregate: recv = segment_sum(edges, receivers)
    (void)hipMemsetAsync(recvB, 0, (size_t)NN * 128 * 4, stream);
    k_scatter_add<<<(EE * 128) / 256, 256, 0, stream>>>(recvB, edges, cr, EE);

    // node MLP
    k_gemm<3, true ><<<NN / 128, 256, 0, stream>>>(nullptr, nodes, recvB, g, nullptr, nullptr,
                                                   wtN[step][0], nB[0], t0, NN, 9);
    k_gemm<0, true ><<<NN / 128, 256, 0, stream>>>(t0, nullptr, nullptr, nullptr, nullptr, nullptr,
                                                   wtN[step][1], nB[1], t1, NN, 4);
    k_gemm<0, false><<<NN / 128, 256, 0, stream>>>(t1, nullptr, nullptr, nullptr, nullptr, nullptr,
                                                   wtN[step][2], nB[2], t0, NN, 4);

    // nodes = LN(nodes + new_nodes);  edges = LN(edges)
    k_layernorm<<<NN / 8, 256, 0, stream>>>(nodes, nodes, t0,
                                            ln_scale + 128 * step, ln_bias + 128 * step, NN);
    k_layernorm<<<EE / 8, 256, 0, stream>>>(edges, edges, nullptr,
                                            ln_scale + 128 * step, ln_bias + 128 * step, EE);

    // dynamic re-wiring
    if (step < 2) {
      k_sqnorm<<<NN / 256, 256, 0, stream>>>(nodes, sqn);
      k_tobf16<<<NN * 128 / 256, 256, 0, stream>>>(nodes, nodesBF, NN * 128);
      k_knn<<<NN / 16, 32, 0, stream>>>(nodesBF, sqn, sndW, rcvW);
      cs = sndW;
      cr = rcvW;
    }
  }

  // ---- decoder ----
  k_gemm<0, true><<<NN / 128, 256, 0, stream>>>(nodes, nullptr, nullptr, nullptr, nullptr, nullptr,
                                                wtDec0, decB[0], t0, NN, 4);
  k_gemm<0, true><<<NN / 128, 256, 0, stream>>>(t0, nullptr, nullptr, nullptr, nullptr, nullptr,
                                                wtDec1, decB[1], t1, NN, 4);
  k_dec_final<<<(NN * 3 + 255) / 256, 256, 0, stream>>>(t1, decW[2], decB[2], (float*)d_out);
}
